// my_model_72851235274998
// MI455X (gfx1250) — matrix operand order repaired and checked by reading, not compiled, not run on hardware
//
#include <hip/hip_runtime.h>

typedef __attribute__((ext_vector_type(16))) __bf16 v16bf;
typedef __attribute__((ext_vector_type(8)))  __bf16 v8bf;
typedef __attribute__((ext_vector_type(8)))  float  v8f;

#define B_   16
#define S_   64
#define E_   256
#define H_   256
#define HLD  264   // padded LDS row stride (floats) to spread banks

// ---------- helpers ----------
__device__ inline v8f v8f_zero() {
    v8f a;
#pragma unroll
    for (int i = 0; i < 8; ++i) a[i] = 0.f;
    return a;
}

__device__ inline v8f wmma_bf16(v16bf a, v16bf b, v8f c) {
    // D = A(16x32 bf16) * B(32x16 bf16) + C(16x16 f32)
    return __builtin_amdgcn_wmma_f32_16x16x32_bf16(
        false, a, false, b, (short)0, c, false, false);
}

__device__ inline float fast_sig(float x) {
    float e = __expf(-x);
    return __builtin_amdgcn_rcpf(1.f + e);
}

__device__ inline float fast_tanh(float x) {
#if __has_builtin(__builtin_amdgcn_tanhf)
    return __builtin_amdgcn_tanhf(x);
#else
    x = fminf(15.f, fmaxf(-15.f, x));
    float e = __expf(-2.f * x);
    return (1.f - e) * __builtin_amdgcn_rcpf(1.f + e);
#endif
}

// Opaque zero offset (in fragments). Laundering an *index* (not the pointer)
// defeats LICM of the loop-invariant weight-fragment loads (which would need
// 500+ VGPRs and spill to scratch) while preserving the pointer's provenance,
// so the loads still lower to global_load_b128 (not flat_load, which would
// dual-count into DScnt and serialize against our LDS traffic).
__device__ inline size_t opaque_zero() {
    size_t z = 0;
    asm volatile("" : "+s"(z));
    return z;
}

// Build a WMMA A-fragment (16x32 bf16) from a row-major f32 matrix (LDS or global).
// A layout (wave32): lane&15 = row M; half-wave selects K base {0,8};
// elems 0..7 -> K kb+0..7, elems 8..15 -> K kb+16..23.
__device__ inline v16bf make_a_frag(const float* m0, int ldr, int kt, int lane) {
    int m = lane & 15;
    const float* r = m0 + m * ldr + kt * 32 + ((lane & 16) ? 8 : 0);
    v16bf a;
#pragma unroll
    for (int j = 0; j < 8; ++j) { a[j] = (__bf16)r[j]; a[j + 8] = (__bf16)r[16 + j]; }
    return a;
}

// ---------- prep: pack W (row-major [N,K_logical], up to 3 column segments)
// into WMMA B-fragment layout: frag = nt*KT + kt, 32 lanes x 16 bf16 contiguous.
// B layout (wave32): lane&15 = column N; half-wave selects K base {0,8};
// elems 0..7 -> K kb+0..7, elems 8..15 -> K kb+16..23.
__global__ __launch_bounds__(256) void pack_b_frags(
    __bf16* __restrict__ dst,
    const float* __restrict__ s0, int ld0,
    const float* __restrict__ s1, int ld1,
    const float* __restrict__ s2, int ld2,
    int k0, int k01, int N, int K)
{
    int gid   = blockIdx.x * 256 + threadIdx.x;
    int nfrag = (N >> 4) * (K >> 5);
    int frag  = gid >> 5;
    if (frag >= nfrag) return;
    int lane = gid & 31;
    int KT = K >> 5;
    int nt = frag / KT, kt = frag - nt * KT;
    int n  = nt * 16 + (lane & 15);
    int kb = kt * 32 + ((lane & 16) ? 8 : 0);
    v16bf a;
#pragma unroll
    for (int j = 0; j < 16; ++j) {
        int kk = kb + (j < 8 ? j : 8 + j);
        const float* sp; int c;
        if (kk < k0)       { sp = s0 + (size_t)n * ld0; c = kk; }
        else if (kk < k01) { sp = s1 + (size_t)n * ld1; c = kk - k0; }
        else               { sp = s2 + (size_t)n * ld2; c = kk - k01; }
        a[j] = (__bf16)sp[c];
    }
    *(v16bf*)(dst + ((size_t)frag * 32 + lane) * 16) = a;
}

// ---------- prep: embedding gather straight into A-fragment layout
// embA[t][kt][lane][16], rows M = batch b, K over E.
__global__ __launch_bounds__(256) void emb_gather(
    __bf16* __restrict__ dst, const int* __restrict__ x,
    const float* __restrict__ embed)
{
    int gid = blockIdx.x * 256 + threadIdx.x;
    if (gid >= S_ * 8 * 32) return;
    int lane = gid & 31;
    int frag = gid >> 5;
    int kt = frag & 7, t = frag >> 3;
    int b = lane & 15;
    int row = x[b * S_ + t];
    const float* src = embed + (size_t)row * E_ + kt * 32 + ((lane & 16) ? 8 : 0);
    v16bf a;
#pragma unroll
    for (int j = 0; j < 8; ++j) { a[j] = (__bf16)src[j]; a[j + 8] = (__bf16)src[16 + j]; }
    *(v16bf*)(dst + ((size_t)frag * 32 + lane) * 16) = a;
}

// ---------- pass 1: shared LSTM scan (persistent, 1 block x 512 = 16 waves)
// g[16,1024] = e_t @ [W_ih]^T + h @ [W_hh]^T + (b_ih+b_hh); gate tiles
// {w, w+16, w+32, w+48} so wave w owns i/f/g/o for h-cols [16w,16w+16).
__global__ __launch_bounds__(512) void shared_lstm(
    const __bf16* __restrict__ embA, const __bf16* __restrict__ WshB,
    const float* __restrict__ b_ih, const float* __restrict__ b_hh,
    __bf16* __restrict__ sharedRt /* [B][S][H] bf16 */,
    __bf16* __restrict__ sharedA  /* [S][8][32][16]  */)
{
    __shared__ float h_lds[16 * HLD];
    int tid = threadIdx.x, lane = tid & 31, w = tid >> 5;
    for (int i = tid; i < 16 * HLD; i += 512) h_lds[i] = 0.f;
    float cst[8];
#pragma unroll
    for (int e = 0; e < 8; ++e) cst[e] = 0.f;
    float bias[4];
#pragma unroll
    for (int q = 0; q < 4; ++q) {
        int n = (w + q * 16) * 16 + (lane & 15);
        bias[q] = b_ih[n] + b_hh[n];
    }
    const v16bf* Af = (const v16bf*)embA;
    __syncthreads();

    for (int t = 0; t < S_; ++t) {
        size_t z = opaque_zero();              // keep weight loads in-loop
        const v16bf* Wf = (const v16bf*)WshB + z;   // frag = nt*16 + kt (K=512)
        v8f acc[4];
#pragma unroll
        for (int q = 0; q < 4; ++q) acc[q] = v8f_zero();
#pragma unroll
        for (int kt = 0; kt < 16; ++kt) {
            v16bf a = (kt < 8) ? Af[((size_t)(t * 8 + kt)) * 32 + lane]
                               : make_a_frag(h_lds, HLD, kt - 8, lane);
#pragma unroll
            for (int q = 0; q < 4; ++q)
                acc[q] = wmma_bf16(a, Wf[(size_t)(w + q * 16) * 16 + kt], acc[q]);
        }
        float hN[8];
#pragma unroll
        for (int e = 0; e < 8; ++e) {
            float gi = acc[0][e] + bias[0], gf = acc[1][e] + bias[1];
            float gg = acc[2][e] + bias[2], go = acc[3][e] + bias[3];
            float cn = fast_sig(gf) * cst[e] + fast_sig(gi) * fast_tanh(gg);
            cst[e] = cn;
            hN[e]  = fast_sig(go) * fast_tanh(cn);
        }
        __syncthreads();                       // all reads of old h done
        int hc = w * 16 + (lane & 15);
#pragma unroll
        for (int e = 0; e < 8; ++e) {
            int m = e + ((lane & 16) ? 8 : 0);
            h_lds[m * HLD + hc] = hN[e];
            sharedRt[((size_t)m * S_ + t) * H_ + hc] = (__bf16)hN[e];
        }
        __syncthreads();                       // new h visible
        if (w < 8) {                           // emit A-fragments of h_t
            v16bf a = make_a_frag(h_lds, HLD, w, lane);
            *(v16bf*)(sharedA + ((size_t)(t * 8 + w) * 32 + lane) * 16) = a;
        }
    }
}

// ---------- pass 2: base[s,b,:] = shared@Ws1^T + emb@Ws3^T + Ws_b  (wide GEMM)
__global__ __launch_bounds__(256) void base_gemm(
    const __bf16* __restrict__ sharedA, const __bf16* __restrict__ embA,
    const __bf16* __restrict__ WbB, const float* __restrict__ Ws_b,
    __bf16* __restrict__ base /* [S][B][H] bf16 */)
{
    int tid = threadIdx.x, lane = tid & 31, w = tid >> 5;
    int g = blockIdx.x * 8 + w;         // 1024 waves total
    int t = g >> 4, nt = g & 15;
    const v16bf* Af = (const v16bf*)sharedA;
    const v16bf* Ef = (const v16bf*)embA;
    const v16bf* Wf = (const v16bf*)WbB;   // frag = nt*16 + kt (K=512)
    v8f acc = v8f_zero();
#pragma unroll
    for (int kt = 0; kt < 8; ++kt)
        acc = wmma_bf16(Af[((size_t)(t * 8 + kt)) * 32 + lane], Wf[(size_t)nt * 16 + kt], acc);
#pragma unroll
    for (int kt = 0; kt < 8; ++kt)
        acc = wmma_bf16(Ef[((size_t)(t * 8 + kt)) * 32 + lane], Wf[(size_t)nt * 16 + 8 + kt], acc);
    int hc = nt * 16 + (lane & 15);
    float bias = Ws_b[hc];
#pragma unroll
    for (int e = 0; e < 8; ++e) {
        int m = e + ((lane & 16) ? 8 : 0);
        base[((size_t)t * 16 + m) * H_ + hc] = (__bf16)(acc[e] + bias);
    }
}

// ---------- pass 3: task LSTM with attention (persistent, 1 block x 1024)
__global__ __launch_bounds__(1024) void task_lstm(
    const __bf16* __restrict__ embA, const __bf16* __restrict__ Ws2B,
    const __bf16* __restrict__ WclB, const __bf16* __restrict__ base,
    const __bf16* __restrict__ sharedRt, const int* __restrict__ mask,
    const float* __restrict__ Us_w, const float* __restrict__ Us_b,
    const float* __restrict__ b_cell, const float* __restrict__ fc_w,
    const float* __restrict__ fc_b, float* __restrict__ out)
{
    __shared__ float h_lds[16 * HLD];
    __shared__ float proj[16 * HLD];
    __shared__ float Rt[16 * HLD];
    __shared__ float Si[16 * 68];
    __shared__ float att[16 * 68];
    __shared__ float usw[H_];
    int tid = threadIdx.x, lane = tid & 31, w = tid >> 5;   // w: 0..31
    for (int i = tid; i < 16 * HLD; i += 1024) { h_lds[i] = 0.f; proj[i] = 0.f; Rt[i] = 0.f; }
    if (tid < H_) usw[tid] = Us_w[tid];
    float usb = Us_b[0];
    float cst[8];
#pragma unroll
    for (int e = 0; e < 8; ++e) cst[e] = 0.f;
    float bias[4] = {0.f, 0.f, 0.f, 0.f};
    if (w < 16) {
#pragma unroll
        for (int q = 0; q < 4; ++q) {
            int n = (w + q * 16) * 16 + (lane & 15);
            bias[q] = b_cell[n];
        }
    }
    const v16bf* Ef = (const v16bf*)embA;
    __syncthreads();

    for (int t = 0; t < S_; ++t) {
        size_t z = opaque_zero();               // keep weight loads in-loop
        const v16bf* W2f = (const v16bf*)Ws2B + z;   // frag = nt*8 + kt  (K=256)
        const v16bf* Wcf = (const v16bf*)WclB + z;   // frag = nt*24 + kt (K=768)
        // phase 1: proj = h @ Ws2^T   (waves 0..15)
        if (w < 16) {
            v8f a = v8f_zero();
#pragma unroll
            for (int kt = 0; kt < 8; ++kt)
                a = wmma_bf16(make_a_frag(h_lds, HLD, kt, lane), W2f[(size_t)w * 8 + kt], a);
            int hc = w * 16 + (lane & 15);
#pragma unroll
            for (int e = 0; e < 8; ++e) {
                int m = e + ((lane & 16) ? 8 : 0);
                proj[m * HLD + hc] = a[e];
            }
        }
        __syncthreads();
        // phase 2: Si[b,s] = sum_h tanh(base+proj)*u + ub  (one (b,s) per thread)
        {
            int b = tid & 15, s = tid >> 4;
            const v8bf* br8 = (const v8bf*)(base + ((size_t)s * 16 + b) * H_);
            const float* pr = proj + b * HLD;
            float sum = 0.f;
            for (int c = 0; c < H_ / 8; ++c) {
                v8bf v = br8[c];
#pragma unroll
                for (int j = 0; j < 8; ++j)
                    sum += fast_tanh((float)v[j] + pr[c * 8 + j]) * usw[c * 8 + j];
            }
            Si[b * 68 + s] = sum + usb;
        }
        __syncthreads();
        // phase 3: masked softmax over S (wave w handles b=w)
        if (w < 16) {
            int b = w;
            float v0 = Si[b * 68 + lane], v1 = Si[b * 68 + 32 + lane];
            if (mask[b * S_ + lane] == 0)      v0 = -1e9f;
            if (mask[b * S_ + 32 + lane] == 0) v1 = -1e9f;
            float mx = fmaxf(v0, v1);
#pragma unroll
            for (int o = 16; o > 0; o >>= 1) mx = fmaxf(mx, __shfl_xor(mx, o));
            float e0 = __expf(v0 - mx), e1 = __expf(v1 - mx);
            float sm = e0 + e1;
#pragma unroll
            for (int o = 16; o > 0; o >>= 1) sm += __shfl_xor(sm, o);
            float inv = __builtin_amdgcn_rcpf(sm);
            att[b * 68 + lane] = e0 * inv;
            att[b * 68 + 32 + lane] = e1 * inv;
        }
        __syncthreads();
        // phase 4: Rt[b,h] = sum_s att[b,s]*shared[b,s,h]  (4 (b,h) per thread)
        {
            int hh = tid & 255, b0 = tid >> 8;
#pragma unroll
            for (int q = 0; q < 4; ++q) {
                int b = b0 + q * 4;
                const __bf16* sr = sharedRt + (size_t)b * S_ * H_ + hh;
                float sum = 0.f;
                for (int s = 0; s < S_; ++s) sum += att[b * 68 + s] * (float)sr[(size_t)s * H_];
                Rt[b * HLD + hh] = sum;
            }
        }
        __syncthreads();
        // phase 5: gates = e_t@Wx^T + h@Wh^T + Rt@Wm^T + b_cell  (waves 0..15)
        float hN[8] = {0.f, 0.f, 0.f, 0.f, 0.f, 0.f, 0.f, 0.f};
        if (w < 16) {
            v8f acc[4];
#pragma unroll
            for (int q = 0; q < 4; ++q) acc[q] = v8f_zero();
#pragma unroll
            for (int kt = 0; kt < 24; ++kt) {
                v16bf a;
                if (kt < 8)       a = Ef[((size_t)(t * 8 + kt)) * 32 + lane];
                else if (kt < 16) a = make_a_frag(h_lds, HLD, kt - 8, lane);
                else              a = make_a_frag(Rt, HLD, kt - 16, lane);
#pragma unroll
                for (int q = 0; q < 4; ++q)
                    acc[q] = wmma_bf16(a, Wcf[(size_t)(w + q * 16) * 24 + kt], acc[q]);
            }
#pragma unroll
            for (int e = 0; e < 8; ++e) {
                float gi = acc[0][e] + bias[0], gf = acc[1][e] + bias[1];
                float gg = acc[2][e] + bias[2], go = acc[3][e] + bias[3];
                float cn = fast_sig(gf) * cst[e] + fast_sig(gi) * fast_tanh(gg);
                cst[e] = cn;
                hN[e]  = fast_sig(go) * fast_tanh(cn);
            }
        }
        __syncthreads();
        if (w < 16) {
            int hc = w * 16 + (lane & 15);
#pragma unroll
            for (int e = 0; e < 8; ++e) {
                int m = e + ((lane & 16) ? 8 : 0);
                h_lds[m * HLD + hc] = hN[e];
            }
        }
        __syncthreads();
    }
    // final classifier: out[b] = sigmoid(h_last[b,:] . fc_w + fc_b)
    if (w < 16) {
        int b = w;
        float s = 0.f;
        for (int hh = lane; hh < H_; hh += 32) s += h_lds[b * HLD + hh] * fc_w[hh];
#pragma unroll
        for (int o = 16; o > 0; o >>= 1) s += __shfl_xor(s, o);
        if (lane == 0) out[b] = fast_sig(s + fc_b[0]);
    }
}

// ---------- host ----------
extern "C" void kernel_launch(void* const* d_in, const int* in_sizes, int n_in,
                              void* d_out, int out_size, void* d_ws, size_t ws_size,
                              hipStream_t stream)
{
    const int*   x      = (const int*)d_in[0];
    const int*   mask   = (const int*)d_in[1];
    const float* embed  = (const float*)d_in[3];
    const float* W_ih   = (const float*)d_in[4];
    const float* W_hh   = (const float*)d_in[5];
    const float* b_ih   = (const float*)d_in[6];
    const float* b_hh   = (const float*)d_in[7];
    const float* Ws_w   = (const float*)d_in[8];
    const float* Ws_b   = (const float*)d_in[9];
    const float* Us_w   = (const float*)d_in[10];
    const float* Us_b   = (const float*)d_in[11];
    const float* Wx     = (const float*)d_in[12];
    const float* Wh     = (const float*)d_in[13];
    const float* Wm     = (const float*)d_in[14];
    const float* b_cell = (const float*)d_in[15];
    const float* fc_w   = (const float*)d_in[16];
    const float* fc_b   = (const float*)d_in[17];

    char* ws = (char*)d_ws;
    __bf16* embA  = (__bf16*)(ws + 0);        //  512 KB: emb A-frags [S][8][32][16]
    __bf16* shA   = (__bf16*)(ws + 524288);   //  512 KB: shared A-frags
    __bf16* shRt  = (__bf16*)(ws + 1048576);  //  512 KB: shared [B][S][H] bf16
    __bf16* baseB = (__bf16*)(ws + 1572864);  //  512 KB: base [S][B][H] bf16
    __bf16* WshB  = (__bf16*)(ws + 2097152);  // 1024 KB: [W_ih|W_hh] B-frags N=1024 K=512
    __bf16* WbsB  = (__bf16*)(ws + 3145728);  //  256 KB: Ws(shared,emb cols) N=256 K=512
    __bf16* Ws2B  = (__bf16*)(ws + 3407872);  //  128 KB: Ws(h cols) N=256 K=256
    __bf16* WclB  = (__bf16*)(ws + 3538944);  // 1536 KB: [Wx|Wh|Wm] B-frags N=1024 K=768

    // weight packing (wide, parallel)
    pack_b_frags<<<128, 256, 0, stream>>>(WshB, W_ih, 256, W_hh, 256, W_ih, 256,
                                          256, 512, 1024, 512);
    pack_b_frags<<<32, 256, 0, stream>>>(WbsB, Ws_w, 768, Ws_w + 512, 768, Ws_w, 768,
                                         256, 512, 256, 512);
    pack_b_frags<<<16, 256, 0, stream>>>(Ws2B, Ws_w + 256, 768, Ws_w, 768, Ws_w, 768,
                                         256, 256, 256, 256);
    pack_b_frags<<<192, 256, 0, stream>>>(WclB, Wx, 256, Wh, 256, Wm, 256,
                                          256, 512, 1024, 768);
    // embedding gather into A-fragment layout
    emb_gather<<<64, 256, 0, stream>>>(embA, x, embed);
    // pass 1: shared LSTM scan
    shared_lstm<<<1, 512, 0, stream>>>(embA, WshB, b_ih, b_hh, shRt, shA);
    // pass 2: precompute attention base (step-invariant part)
    base_gemm<<<128, 256, 0, stream>>>(shA, embA, WbsB, Ws_b, baseB);
    // pass 3: task LSTM + attention + classifier
    task_lstm<<<1, 1024, 0, stream>>>(embA, Ws2B, WclB, baseB, shRt, mask,
                                      Us_w, Us_b, b_cell, fc_w, fc_b, (float*)d_out);

    (void)in_sizes; (void)n_in; (void)out_size; (void)ws_size;
}
